// ComplexFrameLayer_44117904065145
// MI455X (gfx1250) — compile-verified, matrix operand order checked by reading, not run on hardware
//
#include <hip/hip_runtime.h>

// ---------------------------------------------------------------------------
// ComplexFrameLayer fused kernel for gfx1250 (MI455X)
//   GEMMs via v_wmma_f32_16x16x32_bf16, weight staging via TDM
//   (tensor_load_to_lds + s_wait_tensorcnt), scalarization via VALU,
//   scatter via global_atomic_add_f32, lig_vec warmed via global_prefetch.
// ---------------------------------------------------------------------------

#define DD      128     // feature dim
#define TILE_E  128     // edges per block
#define NTHREADS 256    // 8 waves of 32

typedef __attribute__((ext_vector_type(16))) __bf16        v16bf;
typedef __attribute__((ext_vector_type(8)))  float         v8f;
typedef __attribute__((ext_vector_type(4)))  float         v4f;
typedef __attribute__((ext_vector_type(4)))  unsigned int  v4u;
typedef __attribute__((ext_vector_type(8)))  unsigned int  v8u;

union Frag16 { unsigned int u[8]; v4u q[2]; v16bf b; };

__device__ __forceinline__ unsigned int f2bf_bits(float f) {
  unsigned int u = __float_as_uint(f);
  return (u + 0x7FFFu + ((u >> 16) & 1u)) >> 16;   // round-to-nearest-even
}
__device__ __forceinline__ unsigned int pack2bf(float lo, float hi) {
  return f2bf_bits(lo) | (f2bf_bits(hi) << 16);
}
__device__ __forceinline__ float silu_f(float x) {
  return x / (1.0f + __expf(-x));
}

// ---------------------------------------------------------------------------
// Zero the output (atomically accumulated later).
// ---------------------------------------------------------------------------
__global__ void zero_out_kernel(float* __restrict__ out, int n) {
  int i = blockIdx.x * blockDim.x + threadIdx.x;
  if (i < n) out[i] = 0.0f;
}

// ---------------------------------------------------------------------------
// Pack W1 / W2 (f32 row-major [K=128][N=128]) into per-lane WMMA B-fragment
// layout, bf16 pairs.  dst layout (u32):
//   idx = which*8192 + nt*1024 + kt*256 + lane*8 + v
// lane: n = nt*16 + lane%16, half = lane/16
// vgpr v: k_local = (v<4 ? 2v : 16+2(v-4)) + half*8 ; u32 = {bf16(k),bf16(k+1)}
// ---------------------------------------------------------------------------
__global__ void pack_weights_kernel(const float* __restrict__ W1,
                                    const float* __restrict__ W2,
                                    unsigned int* __restrict__ dst) {
  int idx = blockIdx.x * blockDim.x + threadIdx.x;   // 0 .. 16383
  if (idx >= 16384) return;
  int v     =  idx        & 7;
  int lane  = (idx >> 3)  & 31;
  int kt    = (idx >> 8)  & 3;
  int nt    = (idx >> 10) & 7;
  int which = (idx >> 13) & 1;
  const float* W = which ? W2 : W1;
  int n  = nt * 16 + (lane & 15);
  int hf = lane >> 4;
  int kl = (v < 4 ? 2 * v : 16 + 2 * (v - 4)) + hf * 8;
  int k  = kt * 32 + kl;
  unsigned int lo = f2bf_bits(W[(size_t)k * DD + n]);
  unsigned int hi = f2bf_bits(W[(size_t)(k + 1) * DD + n]);
  dst[idx] = lo | (hi << 16);
}

// ---------------------------------------------------------------------------
// Fused edge kernel.
// ---------------------------------------------------------------------------
__global__ __launch_bounds__(NTHREADS, 2) void fused_edge_kernel(
    const float* __restrict__ h_lig,  const float* __restrict__ h_res,
    const int*   __restrict__ eli,    const int*   __restrict__ eri,
    const float* __restrict__ lig_vec,const float* __restrict__ ediff,
    const float* __restrict__ frame,  const unsigned int* __restrict__ Wpk,
    const float* __restrict__ inb1,   const float* __restrict__ inb2,
    const float* __restrict__ s1W1,   const float* __restrict__ s1b1,
    const float* __restrict__ s1W2,   const float* __restrict__ s1b2,
    const float* __restrict__ s2W1,   const float* __restrict__ s2b1,
    const float* __restrict__ s2W2,   const float* __restrict__ s2b2,
    float* __restrict__ out, int nE)
{
  extern __shared__ unsigned int sm[];
  unsigned int* ldsW  = sm;                      // 16384 u32 : W1,W2 fragments
  unsigned int* hA    = sm + 16384;              //  8192 u32 : 128x128 bf16 tile
  float* u1     = (float*)(sm + 24576);          //  4096 f32
  float* u2     = u1 + 4096;                     //  4096 f32
  float* proj0s = u2 + 4096;                     //   128 f32
  float* difs   = proj0s + 128;                  //   384 f32
  int*   lis    = (int*)(difs + 384);            //   128 i32
  float* cs     = (float*)(lis + 128);           //   128 f32 (b1s,w2s,b1s2,w2s2)

  const int tid  = threadIdx.x;
  const int wave = tid >> 5;
  const int lane = tid & 31;
  const int hf   = lane >> 4;
  const int l16  = lane & 15;
  const int e0   = blockIdx.x * TILE_E;

  // ---- stage packed weights (64 KB) via Tensor Data Mover, wave 0 only ----
  if (wave == 0) {
    // D# group 0: count=1 | lds_addr[63:32] | global_addr[120:64] | type=2
    unsigned long long ga = (unsigned long long)(uintptr_t)Wpk;
    unsigned int ldsbyte  = (unsigned int)(uintptr_t)ldsW;   // LDS byte offset
    v4u g0;
    g0.x = 1u;                                            // count=1, user mode
    g0.y = ldsbyte;                                       // lds_addr
    g0.z = (unsigned int)(ga & 0xFFFFFFFFu);              // global_addr lo
    g0.w = (unsigned int)((ga >> 32) & 0x01FFFFFFu) | (2u << 30);  // hi | type=2
    // D# group 1: mask=0, data_size=3 (8B), 1 row of 8192 elements (64 KB)
    v8u g1;
    g1.s0 = (3u << 16);            // workgroup_mask=0, data_size=8B, no flags
    g1.s1 = (8192u & 0xFFFFu) << 16;        // tensor_dim0[15:0] at bits 63:48
    g1.s2 = (8192u >> 16) | (1u << 16);     // tensor_dim0[31:16] | tensor_dim1 lo
    g1.s3 = (8192u << 16);                  // tensor_dim1 hi=0 | tile_dim0=8192
    g1.s4 = 1u;                             // tile_dim1=1, tile_dim2=0
    g1.s5 = 8192u;                          // tensor_dim0_stride lo
    g1.s6 = 0u;                             // stride0 hi=0, stride1 lo=0
    g1.s7 = 0u;                             // stride1 hi=0
    asm volatile("tensor_load_to_lds %0, %1" :: "s"(g0), "s"(g1) : "memory");
    __builtin_amdgcn_s_wait_tensorcnt(0);
  }

  // ---- stage small scalarization constants ----
  if (tid < 32)        cs[tid] = s1b1[tid];
  else if (tid < 64)   cs[tid] = s1W2[tid - 32];
  else if (tid < 96)   cs[tid] = s2b1[tid - 64];
  else if (tid < 128)  cs[tid] = s2W2[tid - 96];

  // ---- per-wave gather: 16 edges, h = h_lig[li] + h_res[ri] -> bf16 LDS ----
  {
    int erow = wave * 16 + (lane >> 1);          // 2 lanes per edge row
    int e = e0 + erow; if (e >= nE) e = nE - 1;
    int a = eli[e], b = eri[e];
    if ((lane & 1) == 0) lis[erow] = a;
    // warm the lig_vec rows the epilogue will read (global_prefetch_b8)
    __builtin_prefetch(lig_vec + (size_t)a * 384 + (lane & 1) * 192, 0, 0);
    __builtin_prefetch(lig_vec + (size_t)a * 384 + (lane & 1) * 192 + 96, 0, 0);
    const float* pL = h_lig + (size_t)a * DD + (lane & 1) * 64;
    const float* pR = h_res + (size_t)b * DD + (lane & 1) * 64;
    unsigned int* dp = hA + erow * 64 + (lane & 1) * 32;
#pragma unroll
    for (int c = 0; c < 64; c += 4) {
      v4f x = *(const v4f*)(pL + c);
      v4f y = *(const v4f*)(pR + c);
      dp[c / 2]     = pack2bf(x.x + y.x, x.y + y.y);
      dp[c / 2 + 1] = pack2bf(x.z + y.z, x.w + y.w);
    }
  }

  // ---- per-edge frame projection + u-vectors (rank-1 factorization) ----
  {
    int erow = wave * 16 + l16;                  // both halves same edge
    int e = e0 + erow; if (e >= nE) e = nE - 1;
    int b = eri[e];
    float d0 = ediff[e * 3 + 0], d1 = ediff[e * 3 + 1], d2 = ediff[e * 3 + 2];
    const float* F = frame + (size_t)b * 9;
    float p0 = d0 * F[0] + d1 * F[3] + d2 * F[6];
    float p1 = d0 * F[1] + d1 * F[4] + d2 * F[7];
    float p2 = d0 * F[2] + d1 * F[5] + d2 * F[8];
    if (hf == 0) {
      proj0s[erow] = p0;
      difs[erow * 3 + 0] = d0; difs[erow * 3 + 1] = d1; difs[erow * 3 + 2] = d2;
    }
#pragma unroll
    for (int kk = 0; kk < 16; ++kk) {
      int k = hf * 16 + kk;
      u1[erow * 32 + k] = p0 * s1W1[k] + p1 * s1W1[32 + k] + p2 * s1W1[64 + k];
      u2[erow * 32 + k] = p0 * s2W1[k] + p1 * s2W1[32 + k] + p2 * s2W1[64 + k];
    }
  }

  __syncthreads();   // publish ldsW (TDM) + cs; hA/u strips are wave-private

  // ---- A fragments for this wave's 16x128 strip (bf16, 16x32 per K-tile) ----
  const int abase = (wave * 16 + l16) * 64 + hf * 4;
  Frag16 afrag[4];
#pragma unroll
  for (int kt = 0; kt < 4; ++kt) {
    afrag[kt].q[0] = *(const v4u*)(hA + abase + kt * 16);
    afrag[kt].q[1] = *(const v4u*)(hA + abase + kt * 16 + 8);
  }

  // ================= GEMM 1: mid = silu(h @ W1 + b1) =================
  unsigned short* hA16 = (unsigned short*)hA;
#pragma unroll
  for (int nt = 0; nt < 8; ++nt) {
    v8f acc = {0.f, 0.f, 0.f, 0.f, 0.f, 0.f, 0.f, 0.f};
#pragma unroll
    for (int kt = 0; kt < 4; ++kt) {
      Frag16 bf;
      const unsigned int* wb = ldsW + (nt * 4 + kt) * 256 + lane * 8;
      bf.q[0] = *(const v4u*)wb;
      bf.q[1] = *(const v4u*)(wb + 4);
      acc = __builtin_amdgcn_wmma_f32_16x16x32_bf16(
          false, afrag[kt].b, false, bf.b, (short)0, acc, false, false);
    }
    int nc = nt * 16 + l16;
    float bias = inb1[nc];
#pragma unroll
    for (int r = 0; r < 8; ++r) {
      float x = acc[r] + bias;
      hA16[(wave * 16 + r + hf * 8) * DD + nc] =
          (unsigned short)f2bf_bits(silu_f(x));
    }
  }

  // reload A fragments from mid (same wave wrote them; LDS in-order per wave)
#pragma unroll
  for (int kt = 0; kt < 4; ++kt) {
    afrag[kt].q[0] = *(const v4u*)(hA + abase + kt * 16);
    afrag[kt].q[1] = *(const v4u*)(hA + abase + kt * 16 + 8);
  }

  // ====== GEMM 2 + scalarization MLPs + equivariant scatter ======
  const float sB1 = s1b2[0], sB2 = s2b2[0];
#pragma unroll 1
  for (int nt = 0; nt < 8; ++nt) {
    v8f acc = {0.f, 0.f, 0.f, 0.f, 0.f, 0.f, 0.f, 0.f};
#pragma unroll
    for (int kt = 0; kt < 4; ++kt) {
      Frag16 bf;
      const unsigned int* wb = ldsW + 8192 + (nt * 4 + kt) * 256 + lane * 8;
      bf.q[0] = *(const v4u*)wb;
      bf.q[1] = *(const v4u*)(wb + 4);
      acc = __builtin_amdgcn_wmma_f32_16x16x32_bf16(
          false, afrag[kt].b, false, bf.b, (short)0, acc, false, false);
    }
    int nc = nt * 16 + l16;
    float bias = inb2[nc];
#pragma unroll 1
    for (int r = 0; r < 8; ++r) {
      int mrow = wave * 16 + r + hf * 8;
      int e = e0 + mrow;
      if (e >= nE) continue;
      float h = acc[r] + bias;               // h_e[e, nc] (post input MLP)
      float aw1 = sB1, aw2 = sB2;
      const float* pu1 = u1 + mrow * 32;
      const float* pu2 = u2 + mrow * 32;
#pragma unroll 4
      for (int k = 0; k < 32; ++k) {
        float x1 = fmaf(h, pu1[k], cs[k]);
        aw1 = fmaf(silu_f(x1), cs[32 + k], aw1);
        float x2 = fmaf(h, pu2[k], cs[64 + k]);
        aw2 = fmaf(silu_f(x2), cs[96 + k], aw2);
      }
      float sp0 = h * proj0s[mrow];
      float w1v = aw1 + sp0;
      float w2v = aw2 + sp0;
      int lidx = lis[mrow];
      const float* lv = lig_vec + (size_t)lidx * 384 + nc;
      float* op = out + (size_t)lidx * 384 + nc;
      float d0 = difs[mrow * 3 + 0];
      float d1 = difs[mrow * 3 + 1];
      float d2 = difs[mrow * 3 + 2];
      atomicAdd(op,       fmaf(lv[0],   w1v, d0 * w2v));
      atomicAdd(op + 128, fmaf(lv[128], w1v, d1 * w2v));
      atomicAdd(op + 256, fmaf(lv[256], w1v, d2 * w2v));
    }
  }
}

// ---------------------------------------------------------------------------
extern "C" void kernel_launch(void* const* d_in, const int* in_sizes, int n_in,
                              void* d_out, int out_size, void* d_ws, size_t ws_size,
                              hipStream_t stream) {
  const float* h_lig   = (const float*)d_in[0];
  const int*   eli     = (const int*)  d_in[3];
  const float* lig_vec = (const float*)d_in[4];
  const float* h_res   = (const float*)d_in[5];
  const int*   eri     = (const int*)  d_in[8];
  const float* ediff   = (const float*)d_in[9];
  const float* frame   = (const float*)d_in[10];
  const float* inW1    = (const float*)d_in[11];
  const float* inb1    = (const float*)d_in[12];
  const float* inW2    = (const float*)d_in[13];
  const float* inb2    = (const float*)d_in[14];
  const float* s1W1    = (const float*)d_in[15];
  const float* s1b1    = (const float*)d_in[16];
  const float* s1W2    = (const float*)d_in[17];
  const float* s1b2    = (const float*)d_in[18];
  const float* s2W1    = (const float*)d_in[19];
  const float* s2b1    = (const float*)d_in[20];
  const float* s2W2    = (const float*)d_in[21];
  const float* s2b2    = (const float*)d_in[22];
  float* out = (float*)d_out;
  unsigned int* Wpk = (unsigned int*)d_ws;      // 64 KB packed bf16 weights
  const int nE = in_sizes[3];                   // 200000 edges

  zero_out_kernel<<<(out_size + NTHREADS - 1) / NTHREADS, NTHREADS, 0, stream>>>(out, out_size);
  pack_weights_kernel<<<16384 / NTHREADS, NTHREADS, 0, stream>>>(inW1, inW2, Wpk);

  int nBlocks = (nE + TILE_E - 1) / TILE_E;
  size_t smem = (size_t)(16384 + 8192 + 4096 + 4096 + 128 + 384 + 128 + 128) * 4;
  fused_edge_kernel<<<nBlocks, NTHREADS, smem, stream>>>(
      h_lig, h_res, eli, eri, lig_vec, ediff, frame, Wpk,
      inb1, inb2, s1W1, s1b1, s1W2, s1b2, s2W1, s2b1, s2W2, s2b2,
      out, nE);
}
